// HOGGraphNet_44375602102649
// MI455X (gfx1250) — compile-verified, matrix operand order checked by reading, not run on hardware
//
#include <hip/hip_runtime.h>
#include <hip/hip_bf16.h>

// ---------------------------------------------------------------------------
// HOG GraphNet (2x GCN + LN + ReLU, 4-head GAT, mean/max graph pooling)
// CDNA5 / gfx1250: wave32, v_wmma_f32_16x16x32_f16 for all dense GEMMs.
// GEMM: one wave owns a 16x128 row-panel -> A fragments built once,
// reused across 8 column tiles (32 WMMAs per wave).
// ---------------------------------------------------------------------------

#define NN 50000
#define EE 600000
#define DD 128
#define HEADS 4
#define GG 64
#define KT 4            // 128 / 32 K-tiles
#define NCT 8           // 128 / 16 column tiles per GEMM call
#define MT (NN / 16)    // 3125 row tiles

typedef __attribute__((ext_vector_type(16))) _Float16 v16h;
typedef __attribute__((ext_vector_type(8)))  float    v8f;

#define CDIV(a, b) (((a) + (b) - 1) / (b))

// ---------------- helpers ---------------------------------------------------

__device__ __forceinline__ void atomicMaxF(float* addr, float v) {
    // total order on float bit patterns without a CAS loop
    if (v >= 0.0f) atomicMax((int*)addr, __float_as_int(v));
    else           atomicMin((unsigned int*)addr, __float_as_uint(v));
}

__device__ __forceinline__ float lrelu(float e) { return e > 0.0f ? e : 0.2f * e; }

// ---------------- generic fills ---------------------------------------------

__global__ void fill_f32(float* __restrict__ p, float v, long long n) {
    long long t = (long long)blockIdx.x * blockDim.x + threadIdx.x;
    if (t < n) p[t] = v;
}

__global__ void init_graph_out(float* __restrict__ gout) {
    int t = blockIdx.x * blockDim.x + threadIdx.x;   // G*256
    if (t >= GG * 2 * DD) return;
    int d = t & 255;
    gout[t] = (d < DD) ? 0.0f : -INFINITY;
}

// ---------------- degree ----------------------------------------------------

__global__ void deg_edge(const int* __restrict__ dst, float* __restrict__ deg) {
    int t = blockIdx.x * blockDim.x + threadIdx.x;
    if (t < EE) atomicAdd(&deg[dst[t]], 1.0f);
}

__global__ void deg_to_dis(float* __restrict__ d) {
    int t = blockIdx.x * blockDim.x + threadIdx.x;
    if (t < NN) d[t] = rsqrtf(d[t]);   // deg >= 1 (self loop)
}

// ---------------- weight packing into WMMA B-fragment layout ----------------
// element e of lane l (for tile kt,ct):
//   K = kt*32 + ((e&8)?16:0) + ((l>=16)?8:0) + (e&7),  N = ct*16 + (l&15)
__global__ void pack_w(const float* __restrict__ W, _Float16* __restrict__ out, int Dout) {
    int t = blockIdx.x * blockDim.x + threadIdx.x;
    int nThreads = (Dout >> 4) * KT * 32;
    if (t >= nThreads) return;
    int lane = t & 31;
    int kt   = (t >> 5) & 3;
    int hi   = (lane >> 4) & 1;
    int n    = ((t >> 7) << 4) + (lane & 15);
    _Float16* o = out + (size_t)t * 16;
#pragma unroll
    for (int e = 0; e < 16; ++e) {
        int k = kt * 32 + ((e & 8) ? 16 : 0) + hi * 8 + (e & 7);
        o[e] = (_Float16)W[(size_t)k * Dout + n];
    }
}

// ---------------- WMMA GEMM: C[M x 128] = A[M x 128] * Bp(ctBase..) ---------
// one 16x128 row-panel per wave: A fragments built once, 32 WMMAs
__global__ void gemm_wmma(const float* __restrict__ A, const _Float16* __restrict__ Bp,
                          float* __restrict__ C, int ctBase) {
    int wave = blockIdx.x * (blockDim.x >> 5) + (threadIdx.x >> 5);
    int lane = threadIdx.x & 31;
    if (wave >= MT) return;                  // wave-uniform
    int hi  = (lane >> 4) & 1;
    const float* arow = A + (size_t)(wave * 16 + (lane & 15)) * DD;
    const v16h*  bp   = (const v16h*)Bp;

    // build the four A fragments (K = 0..127) once
    v16h af[KT];
#pragma unroll
    for (int kt = 0; kt < KT; ++kt) {
        int base0 = kt * 32 + hi * 8;
        float4 a0 = *(const float4*)(arow + base0);
        float4 a1 = *(const float4*)(arow + base0 + 4);
        float4 a2 = *(const float4*)(arow + base0 + 16);
        float4 a3 = *(const float4*)(arow + base0 + 20);
        v16h f;
        f[0]  = (_Float16)a0.x; f[1]  = (_Float16)a0.y; f[2]  = (_Float16)a0.z; f[3]  = (_Float16)a0.w;
        f[4]  = (_Float16)a1.x; f[5]  = (_Float16)a1.y; f[6]  = (_Float16)a1.z; f[7]  = (_Float16)a1.w;
        f[8]  = (_Float16)a2.x; f[9]  = (_Float16)a2.y; f[10] = (_Float16)a2.z; f[11] = (_Float16)a2.w;
        f[12] = (_Float16)a3.x; f[13] = (_Float16)a3.y; f[14] = (_Float16)a3.z; f[15] = (_Float16)a3.w;
        af[kt] = f;
    }

    int mbase = wave * 16 + hi * 8;
#pragma unroll
    for (int ctl = 0; ctl < NCT; ++ctl) {
        int ctg = ctBase + ctl;
        v8f c = {};
#pragma unroll
        for (int kt = 0; kt < KT; ++kt) {
            v16h bf = bp[(size_t)(ctg * KT + kt) * 32 + lane];
            c = __builtin_amdgcn_wmma_f32_16x16x32_f16(false, af[kt], false, bf, (short)0, c, false, false);
        }
        int ncol = ctl * 16 + (lane & 15);
#pragma unroll
        for (int i = 0; i < 8; ++i)
            C[(size_t)(mbase + i) * DD + ncol] = c[i];
    }
}

// ---------------- GCN -------------------------------------------------------

__global__ void gcn_self(const float* __restrict__ h, const float* __restrict__ dis,
                         const float* __restrict__ bias, float* __restrict__ agg) {
    int t = blockIdx.x * blockDim.x + threadIdx.x;
    if (t >= NN * DD) return;
    int n = t >> 7, d = t & 127;
    float w = dis[n] * dis[n];
    agg[t] = bias[d] + w * h[t];
}

__global__ void gcn_edge(const float* __restrict__ h, float* __restrict__ agg,
                         const int* __restrict__ src, const int* __restrict__ dst,
                         const float* __restrict__ dis) {
    long long t = (long long)blockIdx.x * blockDim.x + threadIdx.x;
    int e = (int)(t >> 5);
    if (e >= EE) return;
    int j = (int)(t & 31);
    int s = src[e], d0 = dst[e];
    float w = dis[s] * dis[d0];
    const float4 hv = *(const float4*)(h + (size_t)s * DD + j * 4);
    float* ap = agg + (size_t)d0 * DD + j * 4;
    atomicAdd(ap + 0, w * hv.x); atomicAdd(ap + 1, w * hv.y);
    atomicAdd(ap + 2, w * hv.z); atomicAdd(ap + 3, w * hv.w);
}

// LayerNorm + ReLU, in place, one wave per node
__global__ void ln_relu(float* __restrict__ X, const float* __restrict__ g,
                        const float* __restrict__ b) {
    int wave = blockIdx.x * (blockDim.x >> 5) + (threadIdx.x >> 5);
    int lane = threadIdx.x & 31;
    if (wave >= NN) return;
    float* row = X + (size_t)wave * DD;
    float4 v = *(const float4*)(row + lane * 4);
    float s  = v.x + v.y + v.z + v.w;
    float sq = v.x * v.x + v.y * v.y + v.z * v.z + v.w * v.w;
    for (int o = 16; o; o >>= 1) { s += __shfl_xor(s, o, 32); sq += __shfl_xor(sq, o, 32); }
    float mu  = s * (1.0f / DD);
    float var = sq * (1.0f / DD) - mu * mu;
    float inv = rsqrtf(var + 1e-5f);
    float4 gg = *(const float4*)(g + lane * 4);
    float4 bb = *(const float4*)(b + lane * 4);
    v.x = fmaxf((v.x - mu) * inv * gg.x + bb.x, 0.0f);
    v.y = fmaxf((v.y - mu) * inv * gg.y + bb.y, 0.0f);
    v.z = fmaxf((v.z - mu) * inv * gg.z + bb.z, 0.0f);
    v.w = fmaxf((v.w - mu) * inv * gg.w + bb.w, 0.0f);
    *(float4*)(row + lane * 4) = v;
}

// ---------------- GAT (one head at a time) ----------------------------------

__global__ void gat_att(const float* __restrict__ hh, const float* __restrict__ atts,
                        const float* __restrict__ attd, float* __restrict__ as_,
                        float* __restrict__ ad_, float* __restrict__ m) {
    int wave = blockIdx.x * (blockDim.x >> 5) + (threadIdx.x >> 5);
    int lane = threadIdx.x & 31;
    if (wave >= NN) return;
    const float4 hv = *(const float4*)(hh + (size_t)wave * DD + lane * 4);
    const float4 sv = *(const float4*)(atts + lane * 4);
    const float4 dv = *(const float4*)(attd + lane * 4);
    float ss = hv.x * sv.x + hv.y * sv.y + hv.z * sv.z + hv.w * sv.w;
    float dd = hv.x * dv.x + hv.y * dv.y + hv.z * dv.z + hv.w * dv.w;
    for (int o = 16; o; o >>= 1) { ss += __shfl_xor(ss, o, 32); dd += __shfl_xor(dd, o, 32); }
    if (lane == 0) {
        as_[wave] = ss; ad_[wave] = dd;
        m[wave] = lrelu(ss + dd);        // self edge seeds segment max
    }
}

__global__ void gat_edge_max(const int* __restrict__ src, const int* __restrict__ dst,
                             const float* __restrict__ as_, const float* __restrict__ ad_,
                             float* __restrict__ eeh, float* __restrict__ m) {
    int t = blockIdx.x * blockDim.x + threadIdx.x;
    if (t >= EE) return;
    float e = lrelu(as_[src[t]] + ad_[dst[t]]);
    eeh[t] = e;
    atomicMaxF(&m[dst[t]], e);
}

__global__ void gat_den_init(const float* __restrict__ as_, const float* __restrict__ ad_,
                             const float* __restrict__ m, float* __restrict__ den) {
    int t = blockIdx.x * blockDim.x + threadIdx.x;
    if (t >= NN) return;
    den[t] = __expf(lrelu(as_[t] + ad_[t]) - m[t]);   // self-edge contribution
}

__global__ void gat_acc_init(const float* __restrict__ hh, const float* __restrict__ den,
                             float* __restrict__ acc) {
    int t = blockIdx.x * blockDim.x + threadIdx.x;
    if (t >= NN * DD) return;
    acc[t] = den[t >> 7] * hh[t];
}

__global__ void gat_edge_agg(const int* __restrict__ src, const int* __restrict__ dst,
                             const float* __restrict__ hh, const float* __restrict__ eeh,
                             const float* __restrict__ m, float* __restrict__ den,
                             float* __restrict__ acc) {
    long long t = (long long)blockIdx.x * blockDim.x + threadIdx.x;
    int e = (int)(t >> 5);
    if (e >= EE) return;
    int j = (int)(t & 31);
    int s = src[e], d0 = dst[e];
    float ex = __expf(eeh[e] - m[d0]);
    if (j == 0) atomicAdd(&den[d0], ex);
    const float4 hv = *(const float4*)(hh + (size_t)s * DD + j * 4);
    float* ap = acc + (size_t)d0 * DD + j * 4;
    atomicAdd(ap + 0, ex * hv.x); atomicAdd(ap + 1, ex * hv.y);
    atomicAdd(ap + 2, ex * hv.z); atomicAdd(ap + 3, ex * hv.w);
}

__global__ void gat_merge(const float* __restrict__ acc, const float* __restrict__ den,
                          float* __restrict__ node) {
    int t = blockIdx.x * blockDim.x + threadIdx.x;
    if (t >= NN * DD) return;
    node[t] += (1.0f / HEADS) * acc[t] / den[t >> 7];
}

// ---------------- finalize + pooling ----------------------------------------

__global__ void gat_final_pool(float* __restrict__ node, const float* __restrict__ gat_b,
                               const int* __restrict__ batch, float* __restrict__ gout) {
    int t = blockIdx.x * blockDim.x + threadIdx.x;
    if (t >= NN * DD) return;
    int n = t >> 7, d = t & 127;
    float v = fmaxf(node[t] + gat_b[d], 0.0f);
    node[t] = v;
    int g = batch[n];
    atomicAdd(&gout[(size_t)g * 256 + d], v);
    atomicMaxF(&gout[(size_t)g * 256 + DD + d], v);
}

__global__ void cnt_kernel(const int* __restrict__ batch, float* __restrict__ cnt) {
    int t = blockIdx.x * blockDim.x + threadIdx.x;
    if (t < NN) atomicAdd(&cnt[batch[t]], 1.0f);
}

__global__ void pool_fin(float* __restrict__ gout, const float* __restrict__ cnt) {
    int t = blockIdx.x * blockDim.x + threadIdx.x;
    if (t >= GG * DD) return;
    int g = t >> 7, d = t & 127;
    gout[(size_t)g * 256 + d] /= fmaxf(cnt[g], 1.0f);
}

// ---------------------------------------------------------------------------

extern "C" void kernel_launch(void* const* d_in, const int* in_sizes, int n_in,
                              void* d_out, int out_size, void* d_ws, size_t ws_size,
                              hipStream_t stream) {
    const float* x    = (const float*)d_in[0];
    const int*   ei   = (const int*)d_in[1];
    const int*   batch= (const int*)d_in[2];
    const float* W1   = (const float*)d_in[3];
    const float* b1   = (const float*)d_in[4];
    const float* g1   = (const float*)d_in[5];
    const float* be1  = (const float*)d_in[6];
    const float* W2   = (const float*)d_in[7];
    const float* b2   = (const float*)d_in[8];
    const float* g2   = (const float*)d_in[9];
    const float* be2  = (const float*)d_in[10];
    const float* gatW = (const float*)d_in[11];
    const float* attS = (const float*)d_in[12];   // [4,128]
    const float* attD = (const float*)d_in[13];
    const float* gatB = (const float*)d_in[14];

    const int* src = ei;
    const int* dst = ei + EE;

    // workspace carve-up (floats)
    float* dis = (float*)d_ws;                 // N
    float* P   = dis + NN;                     // N*128  (GEMM out)
    float* Q   = P   + (size_t)NN * DD;        // N*128
    float* R   = Q   + (size_t)NN * DD;        // N*128
    float* hh  = R   + (size_t)NN * DD;        // N*128  (per-head GAT hidden)
    float* acc = hh  + (size_t)NN * DD;        // N*128  (per-head accumulator)
    float* as_ = acc + (size_t)NN * DD;        // N
    float* ad_ = as_ + NN;                     // N
    float* m   = ad_ + NN;                     // N
    float* den = m   + NN;                     // N
    float* eeh = den + NN;                     // E
    float* cnt = eeh + EE;                     // G
    _Float16* Wp = (_Float16*)(cnt + GG);      // packed weights (98304 halfs)
    _Float16* Wp1 = Wp;
    _Float16* Wp2 = Wp + 16384;
    _Float16* Wpg = Wp + 32768;

    float* gout = (float*)d_out;               // [G, 256]
    float* node = gout + (size_t)GG * 2 * DD;  // [N, 128]

    const int B = 256;
    const int gemmBlocks = CDIV(MT, 8);        // 8 waves / block

    // pack weights into WMMA fragment layout (f16)
    pack_w<<<CDIV(1024, B), B, 0, stream>>>(W1, Wp1, 128);
    pack_w<<<CDIV(1024, B), B, 0, stream>>>(W2, Wp2, 128);
    pack_w<<<CDIV(4096, B), B, 0, stream>>>(gatW, Wpg, 512);

    // degrees (self loop -> init 1)
    fill_f32<<<CDIV(NN, B), B, 0, stream>>>(dis, 1.0f, NN);
    deg_edge<<<CDIV(EE, B), B, 0, stream>>>(dst, dis);
    deg_to_dis<<<CDIV(NN, B), B, 0, stream>>>(dis);

    // --- GCN layer 1 ---
    gemm_wmma<<<gemmBlocks, B, 0, stream>>>(x, Wp1, P, 0);
    gcn_self<<<CDIV(NN * DD, B), B, 0, stream>>>(P, dis, b1, Q);
    gcn_edge<<<CDIV((long long)EE * 32, B), B, 0, stream>>>(P, Q, src, dst, dis);
    ln_relu<<<CDIV(NN, 8), B, 0, stream>>>(Q, g1, be1);

    // --- GCN layer 2 ---
    gemm_wmma<<<gemmBlocks, B, 0, stream>>>(Q, Wp2, P, 0);
    gcn_self<<<CDIV(NN * DD, B), B, 0, stream>>>(P, dis, b2, R);
    gcn_edge<<<CDIV((long long)EE * 32, B), B, 0, stream>>>(P, R, src, dst, dis);
    ln_relu<<<CDIV(NN, 8), B, 0, stream>>>(R, g2, be2);

    // --- GAT, head by head ---
    fill_f32<<<CDIV((long long)NN * DD, B), B, 0, stream>>>(node, 0.0f, (long long)NN * DD);
    for (int h = 0; h < HEADS; ++h) {
        gemm_wmma<<<gemmBlocks, B, 0, stream>>>(R, Wpg, hh, h * NCT);
        gat_att<<<CDIV(NN, 8), B, 0, stream>>>(hh, attS + h * DD, attD + h * DD, as_, ad_, m);
        gat_edge_max<<<CDIV(EE, B), B, 0, stream>>>(src, dst, as_, ad_, eeh, m);
        gat_den_init<<<CDIV(NN, B), B, 0, stream>>>(as_, ad_, m, den);
        gat_acc_init<<<CDIV(NN * DD, B), B, 0, stream>>>(hh, den, acc);
        gat_edge_agg<<<CDIV((long long)EE * 32, B), B, 0, stream>>>(src, dst, hh, eeh, m, den, acc);
        gat_merge<<<CDIV(NN * DD, B), B, 0, stream>>>(acc, den, node);
    }

    // --- pooling ---
    init_graph_out<<<CDIV(GG * 256, B), B, 0, stream>>>(gout);
    fill_f32<<<1, B, 0, stream>>>(cnt, 0.0f, GG);
    gat_final_pool<<<CDIV(NN * DD, B), B, 0, stream>>>(node, gatB, batch, gout);
    cnt_kernel<<<CDIV(NN, B), B, 0, stream>>>(batch, cnt);
    pool_fin<<<CDIV(GG * DD, B), B, 0, stream>>>(gout, cnt);
}
